// DynamicToepliztMultiheadV3_39470749450858
// MI455X (gfx1250) — compile-verified, matrix operand order checked by reading, not run on hardware
//
#include <hip/hip_runtime.h>

typedef __attribute__((ext_vector_type(16))) __bf16 v16bf;
typedef __attribute__((ext_vector_type(8)))  float  v8f;

#define NSEQ   4096
#define TWO_N  8192
#define UPITCH 8208          /* 8192 + 16 pad -> wrap-free 16-elem fragment reads */
#define NHEAD  16
#define DDIM   64
#define MD     262080        /* (n-1)*dim = 4095*64 */
#define APITCH 72            /* padded LDS row pitch (bf16) -> bank-conflict-free */

// ---------------------------------------------------------------------------
// Kernel 1: x (B,H,N,DIM) f32  ->  xT[h][d][b][j] bf16   (LDS tile transpose)
// ---------------------------------------------------------------------------
__global__ __launch_bounds__(256) void transpose_x_kernel(
    const float* __restrict__ x, __bf16* __restrict__ xT)
{
  __shared__ float tile[64][65];
  const int bid = blockIdx.x;          // b*16*64 + h*64 + jc
  const int jc  = bid & 63;
  const int h   = (bid >> 6) & 15;
  const int b   = bid >> 10;
  const int tid = threadIdx.x;

  const float* xp = x + ((((size_t)b * NHEAD + h) * NSEQ) + (size_t)jc * 64) * DDIM;
  for (int e = tid; e < 4096; e += 256) {
    int j = e >> 6, d = e & 63;
    tile[j][d] = xp[e];
  }
  __syncthreads();
  for (int e = tid; e < 4096; e += 256) {
    int d = e >> 6, j = e & 63;
    xT[(((size_t)h * DDIM + d) * 8 + b) * NSEQ + (size_t)jc * 64 + j] = (__bf16)tile[j][d];
  }
}

// ---------------------------------------------------------------------------
// Kernel 2: DPB MLP -> time-reversed Toeplitz kernel u[h][d][UPITCH] (bf16)
//   u[m] = a[(2n - m) mod 2n];  u[8192+t] = u[t] (t<16) for wrap-free reads.
// ---------------------------------------------------------------------------
__device__ __forceinline__ void dpb_layer(
    const float* actin, float* actout,
    const float* __restrict__ gW, const float* __restrict__ gB,
    const float* __restrict__ gG, const float* __restrict__ gBe,
    __bf16* Abf, __bf16* wT, float* sb, float* sg, float* sbe,
    int Nout, __bf16* u_final, int uidx0, int d)
{
  const int tid    = threadIdx.x;
  const int wave   = tid >> 5;
  const int lane   = tid & 31;
  const int lane16 = lane & 15;
  const int half   = (lane >> 4) << 4;        // 0 or 16 (K half; contiguous k order, same for A and B)

  __syncthreads();                            // protect act/wT from previous layer
  // stage transposed bf16 weights: wT[n][k] = w[k][n]
  for (int e = tid; e < 64 * Nout; e += 256) {
    int k = e / Nout, n = e - k * Nout;
    wT[n * APITCH + k] = (__bf16)gW[e];
  }
  if (tid < 64)   { sg[tid] = gG[tid]; sbe[tid] = gBe[tid]; }
  if (tid < Nout) { sb[tid] = gB[tid]; }
  __syncthreads();

  // LayerNorm (eps 1e-5) + relu -> bf16 A matrix, one thread per sample row
  if (tid < 64) {
    const float* row = actin + tid * 64;
    float m = 0.f;
    for (int f = 0; f < 64; ++f) m += row[f];
    m *= (1.f / 64.f);
    float v = 0.f;
    for (int f = 0; f < 64; ++f) { float t = row[f] - m; v += t * t; }
    float inv = rsqrtf(v * (1.f / 64.f) + 1e-5f);
    for (int f = 0; f < 64; ++f) {
      float t = (row[f] - m) * inv * sg[f] + sbe[f];
      Abf[tid * APITCH + f] = (__bf16)(t > 0.f ? t : 0.f);
    }
  }
  __syncthreads();

  const int T = Nout >> 2;                    // # of 16x16 output tiles (16 or 4)
  for (int t = wave; t < T; t += 8) {
    const int mi = (Nout == 64) ? (t >> 2) : t;
    const int ni = (Nout == 64) ? (t & 3) : 0;
    v8f c = {};
#pragma unroll
    for (int kk = 0; kk < 64; kk += 32) {
      v16bf a = *(const v16bf*)(Abf + (mi * 16 + lane16) * APITCH + kk + half);
      v16bf b = *(const v16bf*)(wT  + (ni * 16 + lane16) * APITCH + kk + half);
      c = __builtin_amdgcn_wmma_f32_16x16x32_bf16(false, a, false, b, (short)0, c,
                                                  false, false);
    }
    const int   n    = ni * 16 + lane16;
    const float bias = sb[n];
    const int   moff = (lane >= 16) ? 8 : 0;
    if (u_final == nullptr) {
#pragma unroll
      for (int r = 0; r < 8; ++r)
        actout[(mi * 16 + moff + r) * 64 + n] = c[r] + bias;
    } else {                                   // final layer: n = head index
      __bf16* ub = u_final + ((size_t)n * DDIM + d) * UPITCH;
      const int mbase = uidx0 + mi * 16 + moff;
#pragma unroll
      for (int r = 0; r < 8; ++r) {
        float vv = c[r] + bias;
        int   mm = mbase + r;
        ub[mm] = (__bf16)vv;
        if (mm < 16) ub[mm + TWO_N] = (__bf16)vv;   // replicate pad region
      }
    }
  }
}

__global__ __launch_bounds__(256) void dpb_mlp_kernel(
    const float* __restrict__ w0,  const float* __restrict__ b0,
    const float* __restrict__ g1,  const float* __restrict__ be1,
    const float* __restrict__ w1,  const float* __restrict__ b1,
    const float* __restrict__ g2,  const float* __restrict__ be2,
    const float* __restrict__ w2,  const float* __restrict__ b2,
    const float* __restrict__ g3,  const float* __restrict__ be3,
    const float* __restrict__ w3,  const float* __restrict__ b3,
    __bf16* __restrict__ u)
{
  __shared__ float  act0[64 * 64];
  __shared__ float  act1[64 * 64];
  __shared__ __bf16 Abf[64 * APITCH];
  __shared__ __bf16 wT[64 * APITCH];
  __shared__ float  sb[64], sg[64], sbe[64];

  const int tid   = threadIdx.x;
  const int bid   = blockIdx.x;       // (TWO_N/64) * DDIM blocks
  const int d     = bid & 63;
  const int uidx0 = (bid >> 6) << 6;  // 64 consecutive u-positions per block

  // layer 0: act0[s][f] = idx(uidx0+s, d) * w0[f] + b0[f]
  for (int e = tid; e < 4096; e += 256) {
    int s = e >> 6, f = e & 63;
    int uidx = uidx0 + s;
    float val;
    if (uidx == 0 || uidx == 4096) {
      val = 0.f;                                    // zero-pad rows of 'a'
    } else if (uidx < 4096) {                       // neg branch (time-reversed)
      int i = (4095 - uidx) * 64 + d;
      val = -(float)(MD - i) * (1.f / (float)MD);
    } else {                                        // pos branch
      int i = (8191 - uidx) * 64 + d;
      val = (float)(i + 1) * (1.f / (float)MD);
    }
    act0[e] = val * w0[f] + b0[f];
  }

  dpb_layer(act0, act1, w1, b1, g1, be1, Abf, wT, sb, sg, sbe, 64, nullptr, 0, 0);
  dpb_layer(act1, act0, w2, b2, g2, be2, Abf, wT, sb, sg, sbe, 64, nullptr, 0, 0);
  dpb_layer(act0, nullptr, w3, b3, g3, be3, Abf, wT, sb, sg, sbe, 16, u, uidx0, d);
}

// ---------------------------------------------------------------------------
// Kernel 3: Toeplitz GEMM.  out[b,h,i,d] = sum_j u[h,d,(j-i)&8191] * xT[h,d,b,j]
//   per (h,d): M=4096, K=4096, N=8 (padded to 16). bf16 WMMA, fp32 accum.
//   4 M-tiles per wave: 1 B-fragment load feeds 4 independent WMMAs.
//   blockIdx.x = h*64+d (d fastest) so concurrent blocks co-fill out cachelines.
// ---------------------------------------------------------------------------
__global__ __launch_bounds__(256) void toeplitz_wmma_kernel(
    const __bf16* __restrict__ u, const __bf16* __restrict__ xT,
    float* __restrict__ out)
{
  const int hd     = blockIdx.x;              // h*64 + d
  const int h      = hd >> 6;
  const int d      = hd & 63;
  const int tid    = threadIdx.x;
  const int wave   = tid >> 5;
  const int lane   = tid & 31;
  const int lane16 = lane & 15;
  const int half   = (lane >> 4) << 4;        // K half: 0 or 16

  const __bf16* up   = u  + (size_t)hd * UPITCH;
  const __bf16* xcol = xT + (size_t)hd * (8 * NSEQ) + (size_t)lane16 * NSEQ;
  const bool    bval = lane16 < 8;            // N columns 8..15 are zero pad

  const int m0 = blockIdx.y * 512 + wave * 64;   // this wave: rows [m0, m0+64)
  const int i0 = m0 + lane16;                    // A row, tile 0

  v8f c0 = {}, c1 = {}, c2 = {}, c3 = {};
  for (int k0 = 0; k0 < NSEQ; k0 += 32) {
    const int kk = k0 + half;
    v16bf b = {};
    if (bval) b = *(const v16bf*)(xcol + kk);
    const int base = (kk - i0) & (TWO_N - 1);      // Toeplitz: tile t = base - 16t
    v16bf a0 = *(const v16bf*)(up + base);
    v16bf a1 = *(const v16bf*)(up + ((base - 16) & (TWO_N - 1)));
    v16bf a2 = *(const v16bf*)(up + ((base - 32) & (TWO_N - 1)));
    v16bf a3 = *(const v16bf*)(up + ((base - 48) & (TWO_N - 1)));
    c0 = __builtin_amdgcn_wmma_f32_16x16x32_bf16(false, a0, false, b, (short)0, c0,
                                                 false, false);
    c1 = __builtin_amdgcn_wmma_f32_16x16x32_bf16(false, a1, false, b, (short)0, c1,
                                                 false, false);
    c2 = __builtin_amdgcn_wmma_f32_16x16x32_bf16(false, a2, false, b, (short)0, c2,
                                                 false, false);
    c3 = __builtin_amdgcn_wmma_f32_16x16x32_bf16(false, a3, false, b, (short)0, c3,
                                                 false, false);
  }

  if (bval) {
    float* ob = out + ((((size_t)lane16 * NHEAD + h) * NSEQ) * DDIM) + d;
    const int moff = (lane >= 16) ? 8 : 0;
    const v8f cc[4] = {c0, c1, c2, c3};
#pragma unroll
    for (int t = 0; t < 4; ++t) {
      const int r0 = m0 + t * 16 + moff;
#pragma unroll
      for (int r = 0; r < 8; ++r)
        ob[(size_t)(r0 + r) * DDIM] = cc[t][r];
    }
  }
}

// ---------------------------------------------------------------------------
extern "C" void kernel_launch(void* const* d_in, const int* in_sizes, int n_in,
                              void* d_out, int out_size, void* d_ws, size_t ws_size,
                              hipStream_t stream) {
  (void)in_sizes; (void)n_in; (void)out_size; (void)ws_size;
  const float* x   = (const float*)d_in[0];
  const float* w0  = (const float*)d_in[1];
  const float* b0  = (const float*)d_in[2];
  const float* g1  = (const float*)d_in[3];
  const float* be1 = (const float*)d_in[4];
  const float* w1  = (const float*)d_in[5];
  const float* b1  = (const float*)d_in[6];
  const float* g2  = (const float*)d_in[7];
  const float* be2 = (const float*)d_in[8];
  const float* w2  = (const float*)d_in[9];
  const float* b2  = (const float*)d_in[10];
  const float* g3  = (const float*)d_in[11];
  const float* be3 = (const float*)d_in[12];
  const float* w3  = (const float*)d_in[13];
  const float* b3  = (const float*)d_in[14];
  float* out = (float*)d_out;

  // workspace carve-out: xT (64 MiB bf16) then u (~16 MiB bf16)
  __bf16* xT = (__bf16*)d_ws;
  __bf16* u  = (__bf16*)((char*)d_ws +
                         (size_t)8 * NHEAD * NSEQ * DDIM * sizeof(__bf16));

  transpose_x_kernel<<<8 * NHEAD * (NSEQ / 64), 256, 0, stream>>>(x, xT);
  dpb_mlp_kernel<<<(TWO_N / 64) * DDIM, 256, 0, stream>>>(
      w0, b0, g1, be1, w1, b1, g2, be2, w2, b2, g3, be3, w3, b3, u);
  dim3 grid(NHEAD * DDIM, NSEQ / 512);
  toeplitz_wmma_kernel<<<grid, 256, 0, stream>>>(u, xT, out);
}
